// MiniMaxLightningAttention_7679401525834
// MI455X (gfx1250) — compile-verified, matrix operand order checked by reading, not run on hardware
//
#include <hip/hip_runtime.h>
#include <hip/hip_bf16.h>

typedef _Float16 v16h __attribute__((ext_vector_type(16)));
typedef float    v8f  __attribute__((ext_vector_type(8)));

static __device__ __forceinline__ v8f vzero() {
    v8f z;
#pragma unroll
    for (int i = 0; i < 8; ++i) z[i] = 0.0f;
    return z;
}

// Load one 16x32 f16 WMMA operand fragment (A layout; B uses the same layout
// with lane<->N) from a row-major array with `stride` halves per row.
// Lane r=lane&15 selects row, kh=lane>>4 selects K-half:
//   elements 0..7  = K (kh*8 .. kh*8+7)
//   elements 8..15 = K (kh*8+16 .. kh*8+23)
static __device__ __forceinline__ v16h load_frag_h(const _Float16* base, int stride, int lane) {
    const int r  = lane & 15;
    const int kh = (lane >> 4) & 1;
    const _Float16* p = base + (long long)r * stride + kh * 8;
    union { uint4 u[2]; v16h v; } t;
    t.u[0] = *(const uint4*)(p);
    t.u[1] = *(const uint4*)(p + 16);
    return t.v;
}

static __device__ __forceinline__ v8f wmma16(v16h a, v16h b, v8f c) {
    return __builtin_amdgcn_wmma_f32_16x16x32_f16(false, a, false, b, (short)0, c, false, false);
}

static __device__ __forceinline__ v16h scale_frag(v16h a, float s) {
    _Float16 hs = (_Float16)s;
#pragma unroll
    for (int i = 0; i < 16; ++i) a[i] *= hs;
    return a;
}

// ---------------------------------------------------------------------------
// Generic tiled GEMM: C(M,N) = epi( A(M,K) @ B(N,K)^T )
// Block tile 128x128, K-step 32, 256 threads = 8 waves (2x4), wave tile 64x32.
// Double-buffered LDS: prefetch next slab into VGPRs before the 8 WMMAs,
// store to the alternate buffer after, one barrier per K-step.
// EPI: 0 = SiLU -> f16, 1 = sigmoid * rmsnorm(attn) -> f16, 2 = raw -> f32
// ---------------------------------------------------------------------------
template <int EPI, bool AF16>
__global__ __launch_bounds__(256) void gemm_f16(
    const void* __restrict__ Ap, const float* __restrict__ Bp, void* __restrict__ Cp,
    int M, int N, int K,
    const float* __restrict__ attn, const float* __restrict__ invrms,
    const float* __restrict__ normw)
{
    __shared__ __align__(16) _Float16 Ash[2][128 * 40];
    __shared__ __align__(16) _Float16 Bsh[2][128 * 40];

    const int lane = threadIdx.x & 31;
    const int wave = threadIdx.x >> 5;
    const int wm = wave >> 2;            // 0..1
    const int wn = wave & 3;             // 0..3
    const int rowBase = blockIdx.y * 128;
    const int colBase = blockIdx.x * 128;
    const int lr = threadIdx.x >> 1;           // 0..127 tile row
    const int lc = (threadIdx.x & 1) * 16;     // 0 or 16

    const long long aRow = (long long)(rowBase + lr) * K + lc;   // A element base
    const long long bRow = (long long)(colBase + lr) * K + lc;   // B element base

    v8f acc[4][2];
#pragma unroll
    for (int mt = 0; mt < 4; ++mt)
#pragma unroll
        for (int nt = 0; nt < 2; ++nt) acc[mt][nt] = vzero();

    const int steps = K >> 5;

    // ---- prologue: stage slab 0 into buffer 0 ----
    {
        union { uint4 u[2]; _Float16 h[16]; } oa, ob;
        if constexpr (AF16) {
            const _Float16* ap = (const _Float16*)Ap + aRow;
            oa.u[0] = ((const uint4*)ap)[0];
            oa.u[1] = ((const uint4*)ap)[1];
        } else {
            const float* ap = (const float*)Ap + aRow;
#pragma unroll
            for (int i = 0; i < 16; ++i) oa.h[i] = (_Float16)ap[i];
        }
        const float* bp = Bp + bRow;
#pragma unroll
        for (int i = 0; i < 16; ++i) ob.h[i] = (_Float16)bp[i];
        *(uint4*)(Ash[0] + lr * 40 + lc)     = oa.u[0];
        *(uint4*)(Ash[0] + lr * 40 + lc + 8) = oa.u[1];
        *(uint4*)(Bsh[0] + lr * 40 + lc)     = ob.u[0];
        *(uint4*)(Bsh[0] + lr * 40 + lc + 8) = ob.u[1];
    }

    for (int s = 0; s < steps; ++s) {
        __syncthreads();
        const int cur = s & 1;
        const bool hasNext = (s + 1) < steps;

        // ---- issue global loads for the NEXT slab (held in VGPRs) ----
        float fa[16], fb[16];
        uint4 ha[2];
        if (hasNext) {
            const int k0 = (s + 1) << 5;
            if constexpr (AF16) {
                const _Float16* ap = (const _Float16*)Ap + aRow + k0;
                ha[0] = ((const uint4*)ap)[0];
                ha[1] = ((const uint4*)ap)[1];
            } else {
                const float* ap = (const float*)Ap + aRow + k0;
#pragma unroll
                for (int i = 0; i < 16; ++i) fa[i] = ap[i];
            }
            const float* bp = Bp + bRow + k0;
#pragma unroll
            for (int i = 0; i < 16; ++i) fb[i] = bp[i];
        }

        // ---- 8 WMMAs from the current buffer ----
        v16h af[4], bf[2];
#pragma unroll
        for (int mt = 0; mt < 4; ++mt)
            af[mt] = load_frag_h(Ash[cur] + (wm * 64 + mt * 16) * 40, 40, lane);
#pragma unroll
        for (int nt = 0; nt < 2; ++nt)
            bf[nt] = load_frag_h(Bsh[cur] + (wn * 32 + nt * 16) * 40, 40, lane);
#pragma unroll
        for (int mt = 0; mt < 4; ++mt)
#pragma unroll
            for (int nt = 0; nt < 2; ++nt)
                acc[mt][nt] = wmma16(af[mt], bf[nt], acc[mt][nt]);

        // ---- convert & store next slab into the alternate buffer ----
        if (hasNext) {
            const int nxt = cur ^ 1;
            union { uint4 u[2]; _Float16 h[16]; } oa, ob;
            if constexpr (AF16) {
                oa.u[0] = ha[0];
                oa.u[1] = ha[1];
            } else {
#pragma unroll
                for (int i = 0; i < 16; ++i) oa.h[i] = (_Float16)fa[i];
            }
#pragma unroll
            for (int i = 0; i < 16; ++i) ob.h[i] = (_Float16)fb[i];
            *(uint4*)(Ash[nxt] + lr * 40 + lc)     = oa.u[0];
            *(uint4*)(Ash[nxt] + lr * 40 + lc + 8) = oa.u[1];
            *(uint4*)(Bsh[nxt] + lr * 40 + lc)     = ob.u[0];
            *(uint4*)(Bsh[nxt] + lr * 40 + lc + 8) = ob.u[1];
        }
    }

    // epilogue (C layout: M = v + 8*(lane>>4), N = lane&15)
#pragma unroll
    for (int mt = 0; mt < 4; ++mt)
#pragma unroll
        for (int nt = 0; nt < 2; ++nt)
#pragma unroll
            for (int v = 0; v < 8; ++v) {
                const int i = rowBase + wm * 64 + mt * 16 + v + ((lane >> 4) << 3);
                const int j = colBase + wn * 32 + nt * 16 + (lane & 15);
                const float x = acc[mt][nt][v];
                if constexpr (EPI == 0) {
                    ((_Float16*)Cp)[(long long)i * N + j] = (_Float16)(x / (1.0f + __expf(-x)));
                } else if constexpr (EPI == 1) {
                    const float g = 1.0f / (1.0f + __expf(-x));
                    const float a = attn[(long long)i * 2048 + j] * invrms[i] * normw[j];
                    ((_Float16*)Cp)[(long long)i * N + j] = (_Float16)(g * a);
                } else {
                    ((float*)Cp)[(long long)i * N + j] = x;
                }
            }
}

// ---------------------------------------------------------------------------
// Lightning attention: one workgroup per (batch, head); 8 waves scan 16
// blocks of 256 tokens. All matmuls via v_wmma_f32_16x16x32_f16.
// ---------------------------------------------------------------------------
__global__ __launch_bounds__(256) void attn_kernel(const _Float16* __restrict__ qkv,
                                                   float* __restrict__ attn)
{
    const int h    = blockIdx.x & 31;
    const int bb   = blockIdx.x >> 5;
    const int lane = threadIdx.x & 31;
    const int wave = threadIdx.x >> 5;
    const int wrow = wave * 32;

    // slope = (1/2^(8/H))^(h+1) * (1 - 0/(31+1e-5) + 1e-5)
    const float slope = exp2f(-0.25f * (float)(h + 1)) * 1.00001f;
    const float bdv   = __expf(-slope * 256.0f);

    __shared__ __align__(16) _Float16 VTs[64 * 264];  // V^T   [e][s]
    __shared__ __align__(16) _Float16 KTs[64 * 264];  // (K*kd)^T [d][s]
    __shared__ __align__(16) _Float16 Ssc[256 * 72];  // decayed S chunk [i][j_local]
    __shared__ __align__(16) float    stF32[64 * 68]; // state^T master [e][d]
    __shared__ __align__(16) _Float16 stF16[64 * 72]; // state^T f16    [e][d]

    for (int i = threadIdx.x; i < 64 * 68; i += 256) stF32[i] = 0.0f;
    for (int i = threadIdx.x; i < 64 * 72; i += 256) stF16[i] = (_Float16)0.0f;

    for (int t = 0; t < 16; ++t) {
        __syncthreads();
        // ---- stage V^T and decay-scaled K^T ----
        {
            const int s = threadIdx.x;
            const long long tok  = (long long)bb * 4096 + t * 256 + s;
            const _Float16* kb = qkv + tok * 6144 + h * 192 + 64;
            const _Float16* vb = kb + 64;
            const float kd = __expf(-slope * (float)(255 - s));
            union { uint4 u[8]; _Float16 hh[64]; } kk, vv;
#pragma unroll
            for (int i = 0; i < 8; ++i) { kk.u[i] = ((const uint4*)kb)[i]; vv.u[i] = ((const uint4*)vb)[i]; }
#pragma unroll
            for (int d = 0; d < 64; ++d) {
                KTs[d * 264 + s] = (_Float16)((float)kk.hh[d] * kd);
                VTs[d * 264 + s] = vv.hh[d];
            }
        }
        __syncthreads();

        // ---- Q fragments straight from global (rows of this wave's band) ----
        v16h qa[2][2];
#pragma unroll
        for (int mt = 0; mt < 2; ++mt)
#pragma unroll
            for (int ks = 0; ks < 2; ++ks) {
                const long long row0 = (long long)bb * 4096 + t * 256 + wrow + mt * 16;
                qa[mt][ks] = load_frag_h(qkv + row0 * 6144 + h * 192 + ks * 32, 6144, lane);
            }

        v8f out[2][4];
#pragma unroll
        for (int mt = 0; mt < 2; ++mt)
#pragma unroll
            for (int et = 0; et < 4; ++et) out[mt][et] = vzero();

        // ---- inter: (Q*qd) @ state  (uses OLD state) ----
        {
            v16h qs[2][2];
#pragma unroll
            for (int mt = 0; mt < 2; ++mt) {
                const float qd = __expf(-slope * (float)(wrow + mt * 16 + (lane & 15) + 1));
                qs[mt][0] = scale_frag(qa[mt][0], qd);
                qs[mt][1] = scale_frag(qa[mt][1], qd);
            }
#pragma unroll
            for (int et = 0; et < 4; ++et) {
                v16h s0 = load_frag_h(stF16 + et * 16 * 72, 72, lane);
                v16h s1 = load_frag_h(stF16 + et * 16 * 72 + 32, 72, lane);
#pragma unroll
                for (int mt = 0; mt < 2; ++mt) {
                    out[mt][et] = wmma16(qs[mt][0], s0, out[mt][et]);
                    out[mt][et] = wmma16(qs[mt][1], s1, out[mt][et]);
                }
            }
        }

        // ---- intra: S = Q K^T, decay, then S @ V (4 key chunks of 64) ----
        for (int kc = 0; kc < 4; ++kc) {
            v8f sacc[2][4];
#pragma unroll
            for (int mt = 0; mt < 2; ++mt)
#pragma unroll
                for (int nt = 0; nt < 4; ++nt) sacc[mt][nt] = vzero();

#pragma unroll
            for (int nt = 0; nt < 4; ++nt) {
                const long long krow = (long long)bb * 4096 + t * 256 + kc * 64 + nt * 16;
                v16h kb0 = load_frag_h(qkv + krow * 6144 + h * 192 + 64, 6144, lane);
                v16h kb1 = load_frag_h(qkv + krow * 6144 + h * 192 + 64 + 32, 6144, lane);
#pragma unroll
                for (int mt = 0; mt < 2; ++mt) {
                    sacc[mt][nt] = wmma16(qa[mt][0], kb0, sacc[mt][nt]);
                    sacc[mt][nt] = wmma16(qa[mt][1], kb1, sacc[mt][nt]);
                }
            }
            // apply causal decay exp(-slope*(i-j)) and spill to LDS as f16
#pragma unroll
            for (int mt = 0; mt < 2; ++mt)
#pragma unroll
                for (int nt = 0; nt < 4; ++nt)
#pragma unroll
                    for (int v = 0; v < 8; ++v) {
                        const int i  = wrow + mt * 16 + v + ((lane >> 4) << 3);
                        const int jl = nt * 16 + (lane & 15);
                        const int dd = i - (kc * 64 + jl);
                        const float w = (dd >= 0) ? __expf(-slope * (float)dd) : 0.0f;
                        Ssc[i * 72 + jl] = (_Float16)(sacc[mt][nt][v] * w);
                    }
            // S chunk @ V chunk (wave reads only its own rows; LDS is in-order)
            v16h sa[2][2];
#pragma unroll
            for (int mt = 0; mt < 2; ++mt)
#pragma unroll
                for (int ks = 0; ks < 2; ++ks)
                    sa[mt][ks] = load_frag_h(Ssc + (wrow + mt * 16) * 72 + ks * 32, 72, lane);
#pragma unroll
            for (int et = 0; et < 4; ++et) {
                v16h v0 = load_frag_h(VTs + et * 16 * 264 + kc * 64, 264, lane);
                v16h v1 = load_frag_h(VTs + et * 16 * 264 + kc * 64 + 32, 264, lane);
#pragma unroll
                for (int mt = 0; mt < 2; ++mt) {
                    out[mt][et] = wmma16(sa[mt][0], v0, out[mt][et]);
                    out[mt][et] = wmma16(sa[mt][1], v1, out[mt][et]);
                }
            }
        }

        // ---- store block output ----
#pragma unroll
        for (int mt = 0; mt < 2; ++mt)
#pragma unroll
            for (int et = 0; et < 4; ++et)
#pragma unroll
                for (int v = 0; v < 8; ++v) {
                    const int i = wrow + mt * 16 + v + ((lane >> 4) << 3);
                    const long long row = (long long)bb * 4096 + t * 256 + i;
                    attn[row * 2048 + h * 64 + et * 16 + (lane & 15)] = out[mt][et][v];
                }

        __syncthreads();
        // ---- state update: state = bd*state + (K*kd)^T @ V  -> stored as [e][d]
        for (int tl = 0; tl < 2; ++tl) {
            const int tile = wave * 2 + tl;
            const int et = tile >> 2, dt = tile & 3;
            v8f acc = vzero();
#pragma unroll
            for (int ks = 0; ks < 8; ++ks) {
                v16h av = load_frag_h(VTs + et * 16 * 264 + ks * 32, 264, lane);
                v16h bk = load_frag_h(KTs + dt * 16 * 264 + ks * 32, 264, lane);
                acc = wmma16(av, bk, acc);
            }
#pragma unroll
            for (int v = 0; v < 8; ++v) {
                const int e = et * 16 + v + ((lane >> 4) << 3);
                const int d = dt * 16 + (lane & 15);
                const float nv = bdv * stF32[e * 68 + d] + acc[v];
                stF32[e * 68 + d] = nv;
                stF16[e * 72 + d] = (_Float16)nv;
            }
        }
    }
}

// ---------------------------------------------------------------------------
// Per-token RMS: invrms[i] = rsqrt(mean(attn[i,:]^2) + 1e-5)
// ---------------------------------------------------------------------------
__global__ __launch_bounds__(256) void rms_kernel(const float* __restrict__ attn,
                                                  float* __restrict__ invrms)
{
    __shared__ float red[256];
    const long long i = blockIdx.x;
    const float* row = attn + i * 2048;
    float s = 0.0f;
    for (int j = threadIdx.x; j < 2048; j += 256) { const float v = row[j]; s += v * v; }
    red[threadIdx.x] = s;
    __syncthreads();
    for (int st = 128; st > 0; st >>= 1) {
        if (threadIdx.x < (unsigned)st) red[threadIdx.x] += red[threadIdx.x + st];
        __syncthreads();
    }
    if (threadIdx.x == 0) invrms[i] = rsqrtf(red[0] / 2048.0f + 1e-5f);
}

// ---------------------------------------------------------------------------
extern "C" void kernel_launch(void* const* d_in, const int* in_sizes, int n_in,
                              void* d_out, int out_size, void* d_ws, size_t ws_size,
                              hipStream_t stream)
{
    (void)in_sizes; (void)n_in; (void)out_size; (void)ws_size;
    const float* hidden = (const float*)d_in[0];
    const float* qkv_w  = (const float*)d_in[1];
    const float* out_w  = (const float*)d_in[2];
    const float* gate_w = (const float*)d_in[3];
    const float* norm_w = (const float*)d_in[4];

    char* ws = (char*)d_ws;
    _Float16* qkv   = (_Float16*)(ws);                                   // 8192*6144 f16
    float*    attn  = (float*)(ws + 100663296ull);                       // 8192*2048 f32
    float*    invr  = (float*)(ws + 100663296ull + 67108864ull);         // 8192 f32
    _Float16* gated = (_Float16*)(ws + 100663296ull + 67108864ull + 32768ull); // 8192*2048 f16

    // 1) qkv = silu(hidden @ qkv_w^T)  -> f16
    gemm_f16<0, false><<<dim3(48, 64), 256, 0, stream>>>(
        hidden, qkv_w, qkv, 8192, 6144, 2048, nullptr, nullptr, nullptr);
    // 2) lightning attention scan -> attn f32
    attn_kernel<<<64, 256, 0, stream>>>(qkv, attn);
    // 3) per-token inverse RMS
    rms_kernel<<<8192, 256, 0, stream>>>(attn, invr);
    // 4) gated = sigmoid(hidden @ gate_w^T) * rmsnorm(attn) -> f16
    gemm_f16<1, false><<<dim3(16, 64), 256, 0, stream>>>(
        hidden, gate_w, gated, 8192, 2048, 2048, attn, invr, norm_w);
    // 5) out = gated @ out_w^T -> f32
    gemm_f16<2, true><<<dim3(16, 64), 256, 0, stream>>>(
        gated, out_w, d_out, 8192, 2048, 2048, nullptr, nullptr, nullptr);
}
